// BondFlow_30786325578181
// MI455X (gfx1250) — matrix operand order repaired and checked by reading, not compiled, not run on hardware
//
#include <hip/hip_runtime.h>
#include <math.h>

// ---------------------------------------------------------------------------
// Types for CDNA5 WMMA (gfx1250, wave32)
// ---------------------------------------------------------------------------
typedef __bf16 bf16_t;
typedef __attribute__((ext_vector_type(16))) __bf16 v16bf;
typedef __attribute__((ext_vector_type(8)))  __bf16 v8bf;
typedef __attribute__((ext_vector_type(8)))  float  v8f;
typedef __attribute__((ext_vector_type(4)))  float  f32x4;

#define DEV __device__ __forceinline__

DEV int imin(int a, int b) { return a < b ? a : b; }

DEV void atomicMaxF(float* addr, float v) {
  if (v >= 0.f) atomicMax((int*)addr, __float_as_int(v));
  else          atomicMin((unsigned int*)addr, __float_as_uint(v));
}

// f32 -> bf16 weight conversion (run once per launch on all weight leaves)
__global__ void k_f2bf(const float* __restrict__ src, bf16_t* __restrict__ dst, long n) {
  long i = (long)blockIdx.x*256 + threadIdx.x;
  if (i < n) dst[i] = (bf16_t)src[i];
}

// ---------------------------------------------------------------------------
// WMMA GEMM:  C[M,N] (+)= A[M,K](f32) * W[N,K]^T(bf16)   (f32 accumulate)
// Requires K % 32 == 0 (host dispatch guarantees). Out-of-range rows/cols are
// CLAMPED on load (garbage lanes masked at store) so the K-loop is branch-free.
// blockDim=256 (8 waves), per-wave tile 16(M) x 64(N), K step 32.
// Per k-step: 4 b128 A loads + 8 cvt, 8 b128 bf16 B loads (no cvt), 4 WMMAs.
// ---------------------------------------------------------------------------
DEV v16bf cvt16(f32x4 a, f32x4 b, f32x4 c, f32x4 d) {
  v16bf r;
  r[0]=(bf16_t)a[0];  r[1]=(bf16_t)a[1];  r[2]=(bf16_t)a[2];  r[3]=(bf16_t)a[3];
  r[4]=(bf16_t)b[0];  r[5]=(bf16_t)b[1];  r[6]=(bf16_t)b[2];  r[7]=(bf16_t)b[3];
  r[8]=(bf16_t)c[0];  r[9]=(bf16_t)c[1];  r[10]=(bf16_t)c[2]; r[11]=(bf16_t)c[3];
  r[12]=(bf16_t)d[0]; r[13]=(bf16_t)d[1]; r[14]=(bf16_t)d[2]; r[15]=(bf16_t)d[3];
  return r;
}
DEV v16bf ldbf16(const bf16_t* p) {            // 16 contiguous bf16 (two b128)
  v8bf lo = *(const v8bf*)p;
  v8bf hi = *(const v8bf*)(p + 8);
  return __builtin_shufflevector(lo, hi, 0,1,2,3,4,5,6,7,8,9,10,11,12,13,14,15);
}

__global__ __launch_bounds__(256)
void k_gemm(const float* __restrict__ A, const bf16_t* __restrict__ B, float* __restrict__ C,
            int M, int N, int K, int lda, int ldb, int ldc,
            const int* __restrict__ aIdx, long aBatch, long cBatch, int accum)
{
  int lane = threadIdx.x & 31;
  int wv   = threadIdx.x >> 5;
  const float* Ab = A + (long)blockIdx.z * aBatch;
  float*       Cb = C + (long)blockIdx.z * cBatch;
  int n16 = lane & 15;
  int hl  = lane >> 4;

  int rowA = imin(blockIdx.y * 128 + wv * 16 + n16, M - 1);     // clamp
  long rr = aIdx ? (long)aIdx[rowA] : (long)rowA;
  const float* aPtr = Ab + rr * (long)lda;

  int colBase = blockIdx.x * 64;
  const bf16_t* bPtr[4];
#pragma unroll
  for (int f=0; f<4; ++f)
    bPtr[f] = B + (long)imin(colBase + f*16 + n16, N - 1) * ldb;  // clamp

  v8f acc[4];
#pragma unroll
  for (int f=0; f<4; ++f)
#pragma unroll
    for (int r=0; r<8; ++r) acc[f][r] = 0.f;

  for (int k0 = 0; k0 < K; k0 += 32) {
    const f32x4* ap1 = (const f32x4*)(aPtr + k0 + 16*hl);
    const f32x4* ap2 = (const f32x4*)(aPtr + k0 + 16*hl + 8);
    f32x4 a0 = ap1[0], a1 = ap1[1], a2 = ap2[0], a3 = ap2[1];
    v16bf b0 = ldbf16(bPtr[0] + k0 + 16*hl);
    v16bf b1 = ldbf16(bPtr[1] + k0 + 16*hl);
    v16bf b2 = ldbf16(bPtr[2] + k0 + 16*hl);
    v16bf b3 = ldbf16(bPtr[3] + k0 + 16*hl);
    v16bf af = cvt16(a0, a1, a2, a3);
    acc[0] = __builtin_amdgcn_wmma_f32_16x16x32_bf16(false, af, false, b0, (short)0, acc[0], false, false);
    acc[1] = __builtin_amdgcn_wmma_f32_16x16x32_bf16(false, af, false, b1, (short)0, acc[1], false, false);
    acc[2] = __builtin_amdgcn_wmma_f32_16x16x32_bf16(false, af, false, b2, (short)0, acc[2], false, false);
    acc[3] = __builtin_amdgcn_wmma_f32_16x16x32_bf16(false, af, false, b3, (short)0, acc[3], false, false);
  }

  int mBase = blockIdx.y*128 + wv*16 + hl*8;
  bool fullM = (blockIdx.y*128 + wv*16 + 16) <= M;
  bool fullN = (colBase + 64) <= N;
  if (fullM && fullN && !accum) {
#pragma unroll
    for (int f=0; f<4; ++f) {
      int c = colBase + f*16 + n16;
#pragma unroll
      for (int r=0; r<8; ++r) Cb[(long)(mBase + r)*ldc + c] = acc[f][r];
    }
  } else {
#pragma unroll
    for (int f=0; f<4; ++f) {
      int c = colBase + f*16 + n16;
      if (c < N) {
#pragma unroll
        for (int r=0; r<8; ++r) {
          int m = mBase + r;
          if (m < M) {
            float* dst = Cb + (long)m*ldc + c;
            float v = acc[f][r];
            if (accum) v += *dst;
            *dst = v;
          }
        }
      }
    }
  }
}

// Fallback for tiny/odd GEMMs (N<16 or K%32!=0): f32 MACs vs bf16 weights
__global__ void k_gemm_small(const float* __restrict__ A, const bf16_t* __restrict__ B,
                             float* __restrict__ C, int M, int N, int K,
                             int lda, int ldb, int ldc, const int* __restrict__ aIdx,
                             long aBatch, long cBatch, int accum)
{
  long i = (long)blockIdx.x*256 + threadIdx.x;
  long total = (long)M*N;
  if (i >= total) return;
  long m = i / N; int n = (int)(i - m*N);
  const float* Ab = A + (long)blockIdx.y*aBatch;
  float*       Cb = C + (long)blockIdx.y*cBatch;
  long rr = aIdx ? (long)aIdx[m] : m;
  const float* a = Ab + rr*(long)lda;
  const bf16_t* b = B + (long)n*ldb;
  float s = 0.f;
  for (int k=0; k<K; ++k) s += a[k]*(float)b[k];
  float* dst = Cb + m*(long)ldc + n;
  if (accum) s += *dst;
  *dst = s;
}

// ---------------------------------------------------------------------------
// Pointwise / graph kernels
// ---------------------------------------------------------------------------
__global__ void k_fill(float* __restrict__ p, long n, float v) {
  long i = (long)blockIdx.x*256 + threadIdx.x; if (i < n) p[i] = v;
}
// (N,V,3) -> (3,N,V)
__global__ void k_tr_vec(const float* __restrict__ in, float* __restrict__ out, long NV) {
  long i = (long)blockIdx.x*256 + threadIdx.x;
  if (i < NV) { out[i]=in[i*3]; out[NV+i]=in[i*3+1]; out[2*NV+i]=in[i*3+2]; }
}
// gaussian smearing + edge expansion (and optional tri-feature tail)
__global__ void k_edge_geom(const int* __restrict__ ia, const int* __restrict__ ib,
    const float* __restrict__ posA, const float* __restrict__ posB,
    float step, int G, int SW, const float* __restrict__ tri, int triW,
    const float* __restrict__ wexp, float* __restrict__ out_sca,
    float* __restrict__ out_vec, int NV, long n, long vstride)
{
  long e = blockIdx.x; if (e >= n) return;
  int t = threadIdx.x;
  long a = ia ? (long)ia[e] : e;
  float vx = posA[a*3+0], vy = posA[a*3+1], vz = posA[a*3+2];
  if (ib) { long b = (long)ib[e]; vx -= posB[b*3+0]; vy -= posB[b*3+1]; vz -= posB[b*3+2]; }
  float d = sqrtf(vx*vx + vy*vy + vz*vz + 1e-12f);
  if (out_sca && t < SW) {
    float val;
    if (t < G) { float df = d - step*(float)t; val = expf((-0.5f/(step*step))*df*df); }
    else val = tri[e*triW + (t - G)];
    out_sca[e*SW + t] = val;
  }
  if (t < NV) {
    float inv = 1.f/(d + 1e-7f);
    float w = wexp[t];
    out_vec[e*NV + t]             = w*vx*inv;
    out_vec[vstride + e*NV + t]   = w*vy*inv;
    out_vec[2*vstride + e*NV + t] = w*vz*inv;
  }
}
__global__ void k_norm3(const float* __restrict__ vh, float* __restrict__ out, long n, long stride) {
  long i = (long)blockIdx.x*256 + threadIdx.x;
  if (i < n) { float a=vh[i], b=vh[i+stride], c=vh[i+2*stride];
               out[i] = sqrtf(a*a+b*b+c*c + 1e-12f); }
}
__global__ void k_gate(float* __restrict__ V, const float* __restrict__ g,
                       const float* __restrict__ bg, int OV, long n, long stride) {
  long i = (long)blockIdx.x*256 + threadIdx.x;
  if (i < n) { float s = 1.f/(1.f+expf(-(g[i]+bg[i%OV])));
               V[i]*=s; V[i+stride]*=s; V[i+2*stride]*=s; }
}
__global__ void k_leaky(float* __restrict__ x, long n, float slope) {
  long i = (long)blockIdx.x*256 + threadIdx.x;
  if (i < n) { float v=x[i]; x[i] = v>=0.f ? v : slope*v; }
}
__global__ void k_vnleaky(float* __restrict__ V, const float* __restrict__ D, long n, long stride) {
  long i = (long)blockIdx.x*256 + threadIdx.x;
  if (i < n) {
    float v0=V[i], v1=V[i+stride], v2=V[i+2*stride];
    float d0=D[i], d1=D[i+stride], d2=D[i+2*stride];
    float dot = v0*d0+v1*d1+v2*d2;
    if (dot < 0.f) {
      float dsq = d0*d0+d1*d1+d2*d2;
      float g = 0.8f * dot/(dsq+1e-6f);
      V[i]=v0-g*d0; V[i+stride]=v1-g*d1; V[i+2*stride]=v2-g*d2;
    }
  }
}
__global__ void k_ys(float* __restrict__ y, const float* __restrict__ ns,
                     const float* __restrict__ bsca, const int* __restrict__ cj, long n) {
  long i = (long)blockIdx.x*256 + threadIdx.x;
  if (i < n) { long e = i >> 8; int o = (int)(i & 255);
               y[i] = ns[(long)cj[e]*256 + o] * (y[i] + bsca[o]); }
}
__global__ void k_yv(float* __restrict__ yv, const float* __restrict__ t2, const float* __restrict__ t3,
                     const float* __restrict__ be2n, const float* __restrict__ bn2e,
                     const float* __restrict__ nv, const int* __restrict__ cj,
                     long n, long stride, long nvStride) {
  long i = (long)blockIdx.x*256 + threadIdx.x;
  if (i < n) {
    long e = i >> 6; int ch = (int)(i & 63);
    float a = t2[i] + be2n[ch];
    float b = t3[i] + bn2e[ch];
    long src = (long)cj[e]*64 + ch;
    for (int c=0;c<3;++c)
      yv[i + c*stride] = a*nv[src + c*nvStride] + b*yv[i + c*stride];
  }
}
__global__ void k_logits1(const float* __restrict__ qs, const float* __restrict__ qv,
                          const float* __restrict__ ks, const float* __restrict__ kv,
                          const int* __restrict__ qi, float* __restrict__ as_, float* __restrict__ av_,
                          long n, long qvStride, long kvStride) {
  long e = (long)blockIdx.x*256 + threadIdx.x; if (e >= n) return;
  long q = qi[e];
  const float* qr = qs + q*256; const float* kr = ks + e*256;
  float s = 0.f;
  for (int o=0;o<256;++o) s += qr[o]*kr[o];
  as_[e] = s * 0.0625f;                 // 1/sqrt(256)
  float v = 0.f;
  for (int c=0;c<3;++c) {
    const float* qvr = qv + (long)c*qvStride + q*64;
    const float* kvr = kv + (long)c*kvStride + e*64;
    for (int i=0;i<64;++i) v += qvr[i]*kvr[i];
  }
  av_[e] = v * 0.07216878364870323f;    // 1/sqrt(192)
}
__global__ void k_logits4(const float* __restrict__ qs, const float* __restrict__ qv,
                          const float* __restrict__ ks, const float* __restrict__ kv,
                          const int* __restrict__ ii, const int* __restrict__ jj,
                          const float* __restrict__ bs, const float* __restrict__ bv,
                          float* __restrict__ ls, float* __restrict__ lv, long n, long vStride) {
  long p = (long)blockIdx.x*256 + threadIdx.x; if (p >= n) return;
  long i = ii[p], j = jj[p];
  for (int h=0; h<4; ++h) {
    float s = 0.f;
    const float* qr = qs + i*256 + h*64; const float* kr = ks + j*256 + h*64;
    for (int d=0; d<64; ++d) s += qr[d]*kr[d];
    ls[p*4+h] = s*0.125f + bs[p*4+h];   // 1/sqrt(64)
    float v = 0.f;
    for (int c=0;c<3;++c) {
      const float* qvr = qv + (long)c*vStride + i*64 + h*16;
      const float* kvr = kv + (long)c*vStride + j*64 + h*16;
      for (int d=0; d<16; ++d) v += qvr[d]*kvr[d];
    }
    lv[p*4+h] = v*0.14433756729740643f + bv[p*4+h];  // 1/sqrt(48)
  }
}
__global__ void k_segmax(const float* __restrict__ l, const int* __restrict__ seg,
                         float* __restrict__ m, int H, long n) {
  long i = (long)blockIdx.x*256 + threadIdx.x;
  if (i < n) { long r = i / H; int h = (int)(i - r*H); atomicMaxF(&m[(long)seg[r]*H + h], l[i]); }
}
__global__ void k_segexp(float* __restrict__ l, const int* __restrict__ seg,
                         const float* __restrict__ m, float* __restrict__ s, int H, long n) {
  long i = (long)blockIdx.x*256 + threadIdx.x;
  if (i < n) { long r = i / H; int h = (int)(i - r*H);
               long o = (long)seg[r]*H + h;
               float e = expf(l[i] - m[o]); l[i] = e; atomicAdd(&s[o], e); }
}
__global__ void k_segnorm(float* __restrict__ l, const int* __restrict__ seg,
                          const float* __restrict__ s, int H, long n) {
  long i = (long)blockIdx.x*256 + threadIdx.x;
  if (i < n) { long r = i / H; int h = (int)(i - r*H);
               l[i] = l[i] / (s[(long)seg[r]*H + h] + 1e-16f); }
}
__global__ void k_scat_s(float* __restrict__ out, const float* __restrict__ vs,
                         const float* __restrict__ w, const int* __restrict__ seg,
                         const int* __restrict__ src, int C, int Hdiv, int H, long n) {
  long i = (long)blockIdx.x*256 + threadIdx.x;
  if (i < n) { long r = i / C; int c = (int)(i - r*C);
               long sr = src ? (long)src[r] : r;
               atomicAdd(&out[(long)seg[r]*C + c], w[r*H + c/Hdiv] * vs[sr*C + c]); }
}
__global__ void k_scat_v(float* __restrict__ out, const float* __restrict__ vv,
                         const float* __restrict__ w, const int* __restrict__ seg,
                         const int* __restrict__ src, int OV, int Hdiv, int H, long n,
                         long oStride, long vStride) {
  long i = (long)blockIdx.x*256 + threadIdx.x;
  if (i < n) {
    long r = i / OV; int c = (int)(i - r*OV);
    long sr = src ? (long)src[r] : r;
    float a = w[r*H + c/Hdiv];
    long oo = (long)seg[r]*OV + c; long vo = sr*OV + c;
    atomicAdd(&out[oo],             a*vv[vo]);
    atomicAdd(&out[oStride + oo],   a*vv[vStride + vo]);
    atomicAdd(&out[2*oStride + oo], a*vv[2*vStride + vo]);
  }
}
__global__ void k_cat_s(float* __restrict__ cat, const float* __restrict__ ya,
                        const float* __restrict__ nsca, const float* __restrict__ s1,
                        const int* __restrict__ iq, const int* __restrict__ jc, long n) {
  long i = (long)blockIdx.x*256 + threadIdx.x;
  if (i < n) {
    long e = i / 768; int c = (int)(i - e*768);
    float v;
    if (c < 256)      v = ya[(long)iq[e]*256 + c];
    else if (c < 512) v = nsca[(long)jc[e]*256 + (c-256)];
    else              v = s1[e*256 + (c-512)];
    cat[i] = v;
  }
}
__global__ void k_cat_v(float* __restrict__ cat, const float* __restrict__ yv,
                        const float* __restrict__ nvT, const float* __restrict__ v1,
                        const int* __restrict__ iq, const int* __restrict__ jc,
                        long n, long catStride, long yStride, long nStride, long vStride) {
  long i = (long)blockIdx.x*256 + threadIdx.x;
  if (i < n) {
    long e = i / 192; int c = (int)(i - e*192);
    for (int comp=0; comp<3; ++comp) {
      float v;
      if (c < 64)       v = yv[(long)comp*yStride + (long)iq[e]*64 + c];
      else if (c < 128) v = nvT[(long)comp*nStride + (long)jc[e]*64 + (c-64)];
      else              v = v1[(long)comp*vStride + e*64 + (c-128)];
      cat[(long)comp*catStride + i] = v;
    }
  }
}
__global__ void k_bvsq(const float* __restrict__ bv, float* __restrict__ out, long n, long stride) {
  long i = (long)blockIdx.x*256 + threadIdx.x;
  if (i < n) { float a=bv[i], b=bv[i+stride], c=bv[i+2*stride]; out[i]=a*a+b*b+c*c; }
}
__global__ __launch_bounds__(256)
void k_ln_s(float* __restrict__ s2, const float* __restrict__ o,
            const float* __restrict__ g, const float* __restrict__ b) {
  int e = blockIdx.x; int t = threadIdx.x;
  __shared__ float red[256];
  long idx = (long)e*256 + t;
  float x = s2[idx] + o[idx];
  red[t] = x; __syncthreads();
  for (int st=128; st>0; st>>=1) { if (t<st) red[t]+=red[t+st]; __syncthreads(); }
  float mean = red[0]*(1.f/256.f); __syncthreads();
  float dx = x - mean;
  red[t] = dx*dx; __syncthreads();
  for (int st=128; st>0; st>>=1) { if (t<st) red[t]+=red[t+st]; __syncthreads(); }
  float var = red[0]*(1.f/256.f);
  s2[idx] = dx*rsqrtf(var+1e-5f)*g[t] + b[t];
}
__global__ __launch_bounds__(192)
void k_ln_v(float* __restrict__ v2, const float* __restrict__ o,
            const float* __restrict__ g, const float* __restrict__ b, long stride) {
  int e = blockIdx.x; int t = threadIdx.x;       // 0..191
  int comp = t / 64; int ch = t - comp*64;
  __shared__ float red[192];
  long idx = (long)comp*stride + (long)e*64 + ch;
  float x = v2[idx] + o[idx];
  red[t] = x; __syncthreads();
  if (t<64) red[t]+=red[t+128]; __syncthreads();
  if (t<64) red[t]+=red[t+64];  __syncthreads();
  for (int st=32; st>0; st>>=1) { if (t<st) red[t]+=red[t+st]; __syncthreads(); }
  float mean = red[0]*(1.f/192.f); __syncthreads();
  float dx = x - mean;
  red[t] = dx*dx; __syncthreads();
  if (t<64) red[t]+=red[t+128]; __syncthreads();
  if (t<64) red[t]+=red[t+64];  __syncthreads();
  for (int st=32; st>0; st>>=1) { if (t<st) red[t]+=red[t+st]; __syncthreads(); }
  float var = red[0]*(1.f/192.f);
  v2[idx] = dx*rsqrtf(var+1e-5f)*g[ch*3+comp] + b[ch*3+comp];
}
__global__ void k_flow(float* __restrict__ z, float* __restrict__ ld,
                       const float* __restrict__ s, const float* __restrict__ t, long n) {
  long i = (long)blockIdx.x*256 + threadIdx.x;
  if (i < n) { float sv = s[i]; z[i] = (z[i] + t[i]) * expf(sv); ld[i] += sv; }
}

// ---------------------------------------------------------------------------
// Host-side orchestration
// ---------------------------------------------------------------------------
struct GVL { const bf16_t *Wg,*Ws,*Wv1,*Wv2; const float* bg; int is,iv,os,ov,hid; };
struct GVP { const bf16_t* Wd; GVL lin; };

static inline dim3 gr1(long n) { return dim3((unsigned)((n + 255) / 256)); }

static void gemm(hipStream_t st, const float* A, const bf16_t* B, float* C,
                 int M, int N, int K, int lda, int ldb, int ldc,
                 const int* aIdx = nullptr, int batch = 1, long aBatch = 0, long cBatch = 0,
                 int accum = 0) {
  if ((K & 31) != 0 || N < 16) {
    dim3 g((unsigned)(((long)M*N + 255)/256), (unsigned)batch, 1);
    k_gemm_small<<<g, dim3(256), 0, st>>>(A, B, C, M, N, K, lda, ldb, ldc,
                                          aIdx, aBatch, cBatch, accum);
  } else {
    dim3 g((unsigned)((N+63)/64), (unsigned)((M+127)/128), (unsigned)batch);
    k_gemm<<<g, dim3(256), 0, st>>>(A, B, C, M, N, K, lda, ldb, ldc,
                                    aIdx, aBatch, cBatch, accum);
  }
}

static void run_gvl(hipStream_t st, const GVL& p, int N,
                    const float* S_in, const float* V_in, float* S_out, float* V_out,
                    float* vh, float* nrm, float* g, bool scalarOnly = false) {
  long NH = (long)N*p.hid, NV = (long)N*p.iv, NO = (long)N*p.ov;
  gemm(st, V_in, p.Wv1, vh, N, p.hid, p.iv, p.iv, p.iv, p.hid, nullptr, 3, NV, NH);
  k_norm3<<<gr1(NH),256,0,st>>>(vh, nrm, NH, NH);
  gemm(st, nrm,  p.Ws,         S_out, N, p.os, p.hid, p.hid, p.hid+p.is, p.os);
  gemm(st, S_in, p.Ws + p.hid, S_out, N, p.os, p.is,  p.is,  p.hid+p.is, p.os,
       nullptr, 1, 0, 0, 1);
  if (scalarOnly) return;
  gemm(st, vh, p.Wv2, V_out, N, p.ov, p.hid, p.hid, p.hid, p.ov, nullptr, 3, NH, NO);
  gemm(st, S_out, p.Wg, g, N, p.ov, p.os, p.os, p.os, p.ov);
  k_gate<<<gr1(NO),256,0,st>>>(V_out, g, p.bg, p.ov, NO, NO);
}
static void run_gvp(hipStream_t st, const GVP& p, int N, const float* S_in, const float* V_in,
                    float* S_out, float* V_out, float* vh, float* nrm, float* g, float* d) {
  run_gvl(st, p.lin, N, S_in, V_in, S_out, V_out, vh, nrm, g);
  long NS = (long)N*p.lin.os, NO = (long)N*p.lin.ov;
  k_leaky<<<gr1(NS),256,0,st>>>(S_out, NS, 0.01f);
  gemm(st, V_out, p.Wd, d, N, p.lin.ov, p.lin.ov, p.lin.ov, p.lin.ov, p.lin.ov,
       nullptr, 3, NO, NO);
  k_vnleaky<<<gr1(NO),256,0,st>>>(V_out, d, NO, NO);
}

extern "C" void kernel_launch(void* const* d_in, const int* in_sizes, int n_in,
                              void* d_out, int out_size, void* d_ws, size_t ws_size,
                              hipStream_t stream) {
  (void)in_sizes; (void)out_size;
  const int NQ=2048, NC=16384, E=16384, EK=65536, PC=32768;
  const long Pp = 131072;
  const long TRANS = 110764032;   // transient floats
  const long BFCAP = 4194304;     // bf16 weight arena capacity (elements)

  // ---- workspace guard & bf16 weight arena ----
  float* W = (float*)d_ws;
  // persistent sizes computed below give oT = 20709376 floats (< oT_const)
  const size_t oT_const = 20840448;
  if (ws_size < (oT_const + (size_t)TRANS + BFCAP/2 + 16) * 4ULL) return;
  bf16_t* bfA = (bf16_t*)(W + oT_const + TRANS);
  size_t bfOff = 0;

  // ---- parameter pytree walk (sorted dict keys, list order), dual mode ----
  int nLeaf = n_in - 12;
  const float* flatP = (nLeaf == 1) ? (const float*)d_in[0] : nullptr;
  size_t pOff = 0; int pCur = 0;
  auto take = [&](size_t n) -> const float* {
    if (flatP) { const float* r = flatP + pOff; pOff += n; return r; }
    return (const float*)d_in[pCur++];
  };
  auto takeW = [&](size_t n) -> const bf16_t* {   // weight leaf -> bf16 arena
    const float* src = take(n);
    bf16_t* dst = bfA + bfOff;
    bfOff += (n + 7) & ~(size_t)7;                // keep 16B alignment
    k_f2bf<<<gr1((long)n),256,0,stream>>>(src, dst, (long)n);
    return dst;
  };
  auto takeGVL = [&](int is,int iv,int os,int ov) -> GVL {
    GVL g; g.is=is; g.iv=iv; g.os=os; g.ov=ov; g.hid = iv>ov ? iv : ov;
    g.Wg  = takeW((size_t)ov*os);
    g.Ws  = takeW((size_t)os*(is+g.hid));
    g.Wv1 = takeW((size_t)g.hid*iv);
    g.Wv2 = takeW((size_t)ov*g.hid);
    g.bg  = take((size_t)ov);
    return g;
  };
  auto takeGVP = [&](int is,int iv,int os,int ov) -> GVP {
    GVP p; p.Wd = takeW((size_t)ov*ov); p.lin = takeGVL(is,iv,os,ov); return p;
  };

  GVL a_bias = takeGVL(256,64,4,4);
  const float* a_bias_wexp = take(64);
  GVL a_k = takeGVL(256,64,256,64);
  const float* ln_s_b = take(256); const float* ln_s_g = take(256);
  const float* ln_v_b = take(192); const float* ln_v_g = take(192);
  GVL a_q = takeGVL(256,64,256,64);
  GVL a_v = takeGVL(256,64,256,64);
  GVP ef0 = takeGVP(768,192,256,64);
  GVL ef1 = takeGVL(256,64,256,64);
  GVP fl_mlp[6]; GVL fl_s[6], fl_t[6];
  for (int i=0;i<6;++i) { fl_mlp[i]=takeGVP(256,64,256,64);
                          fl_s[i]=takeGVL(256,64,4,64); fl_t[i]=takeGVL(256,64,4,64); }
  GVP ne0 = takeGVP(64,64,256,64);
  GVL ne1 = takeGVL(256,64,256,64);
  GVL pa_k = takeGVL(256,64,256,64), pa_out = takeGVL(256,64,256,64);
  GVL pa_q = takeGVL(256,64,256,64), pa_v  = takeGVL(256,64,256,64);
  const bf16_t* We2n = takeW(64*64);  const bf16_t* Wev  = takeW(64*64);
  const bf16_t* Wn2e = takeW(64*256); const bf16_t* Wsca = takeW(256*64);
  const float* be2n = take(64); const float* bn2e = take(64); const float* bsca = take(256);
  GVP msg_edge = takeGVP(64,64,64,64);
  GVL msg_node = takeGVL(256,64,256,64);
  GVL msg_out  = takeGVL(256,64,256,64);
  GVL root     = takeGVL(256,64,256,64);
  const float* w_root = take(64); const float* w_vec = take(64);
  const float* w_vec3 = take(64);

  int ib = flatP ? 1 : pCur;
  const float* z_edge   = (const float*)d_in[ib+0];
  const float* pos_q    = (const float*)d_in[ib+1];
  const int*   eiq      = (const int*)  d_in[ib+2];  // (2,E)
  const float* cpx_pos  = (const float*)d_in[ib+3];
  const float* node_sca = (const float*)d_in[ib+4];
  const float* node_vec = (const float*)d_in[ib+5];
  const int*   knn      = (const int*)  d_in[ib+6];  // (2,EK)
  const float* atom_emb = (const float*)d_in[ib+7];
  const int*   idx_ei   = (const int*)  d_in[ib+8];
  const int*   idx_ej   = (const int*)  d_in[ib+9];
  const int*   tri_idx  = (const int*)  d_in[ib+10]; // (2,P)
  const float* tri_feat = (const float*)d_in[ib+11]; // (P,5)
  const int* knn_qi = knn;     const int* knn_cj = knn + EK;
  const int* e_iq = eiq;       const int* e_jc = eiq + E;
  const int* tri0 = tri_idx;   const int* tri1 = tri_idx + Pp;

  // ---- f32 arena (hand-planned, lifetime-aliased) ----
  const long EKS = (long)EK*256, EKV = 3L*EK*64, EKC = (long)EK*64;
  const long ES  = (long)E*256,  EV  = 3L*E*64,  EC  = (long)E*64;
  size_t o = 0;
  auto A_ = [&](long n) -> float* { float* p = W + o; o += (size_t)n; return p; };
  float* nvT  = A_(3L*NC*64);
  float* ns   = A_((long)NC*256);
  float* nv   = A_(3L*NC*64);
  float* xs   = A_((long)NQ*256);
  float* xv   = A_(3L*NQ*64);
  float* ysca = A_((long)NQ*256);
  float* yvec = A_(3L*NQ*64);
  float* s2   = A_(ES);
  float* v2   = A_(EV);
  float* bS   = A_(Pp*4);
  float* bVq  = A_(Pp*4);
  float* T = W + o;                 // transient base (o < oT_const)

  float* zb = (float*)d_out;        // z  (E,4)
  float* ld = zb + (long)E*4;       // logdet (E,4)

  // ================= PHASE 0/1: PositionEncoder =================
  k_tr_vec<<<gr1((long)NC*64),256,0,stream>>>(node_vec, nvT, (long)NC*64);

  // root_vec + x = gvl(root, atom_emb, root_vec)
  float* rv = T;                                   // (3,NQ,64) in e_sca slot
  k_edge_geom<<<dim3(NQ),dim3(64),0,stream>>>(nullptr,nullptr,pos_q,nullptr,1.f,0,0,
      nullptr,0,w_root,nullptr,rv,64,NQ,(long)NQ*64);
  { float* vh=T+EKC; float* nr=vh+3L*NQ*64; float* gg=nr+(long)NQ*64;
    run_gvl(stream, root, NQ, atom_emb, rv, xs, xv, vh, nr, gg); }
  // node gvl
  { float* vh=T+EKC; float* nr=vh+3L*NC*64; float* gg=nr+(long)NC*64;
    run_gvl(stream, msg_node, NC, node_sca, nvT, ns, nv, vh, nr, gg); }

  // EK kNN geometry
  float* t_esca = T;                 // EKC
  float* t_evec = T + EKC;           // EKV (also ev in-place, later k/v vh temp)
  float* t_es   = t_evec + EKV;      // EKC (later k/v norm temp)
  float* t_vh1  = t_es + EKC;        // EKV
  float* t_nrm1 = t_vh1 + EKV;       // EKC
  float* t_g1   = t_nrm1 + EKC;      // EKC
  float* t_d1   = t_g1 + EKC;        // EKV
  float* t_t1   = t_vh1;             // EKS  (y_s; later kv/vv region)
  float* t_t2   = t_t1 + EKS;        // EKC
  float* t_t3   = t_t2 + EKC;        // EKC
  float* t_t4   = t_t3 + EKC;        // EKV  (y_v; later q slot)
  float* t_vh2  = t_t4 + EKV;        // EKV  (later ks/vs region)
  float* t_nrm2 = t_vh2 + EKV;       // EKC
  float* t_g2   = t_nrm2 + EKC;      // EKC  (later k/v gate)
  float* t_hs   = t_g2 + EKC;        // EKS
  float* t_hv   = t_hs + EKS;        // EKV
  float* t_as   = t_hv + EKV;  float* t_av  = t_as + EK;
  float* t_m1   = t_av + EK;   float* t_m2  = t_m1 + NQ;
  float* t_sm1  = t_m2 + NQ;   float* t_sm2 = t_sm1 + NQ;
  float* t_os   = t_sm2 + NQ;                 // (NQ,256)
  float* t_ov   = t_os + (long)NQ*256;        // (3,NQ,64)
  float* t_ovh  = t_ov + 3L*NQ*64;
  float* t_onrm = t_ovh + 3L*NQ*64;
  float* t_og   = t_onrm + (long)NQ*64;

  k_edge_geom<<<dim3(EK),dim3(64),0,stream>>>(knn_qi,knn_cj,pos_q,cpx_pos,10.f/63.f,64,64,
      nullptr,0,w_vec,t_esca,t_evec,64,EK,EKC);
  // es,ev = gvp(msg.edge, e_sca, e_vec)   (ev in-place over e_vec)
  run_gvp(stream, msg_edge, EK, t_esca, t_evec, t_es, t_evec, t_vh1, t_nrm1, t_g1, t_d1);
  // message combine
  gemm(stream, t_es, Wsca, t_t1, EK, 256, 64, 64, 64, 256);                 // es@Wsca^T
  gemm(stream, t_es, We2n, t_t2, EK, 64, 64, 64, 64, 64);
  gemm(stream, ns, Wn2e, t_t3, EK, 64, 256, 256, 256, 64, knn_cj);          // gathered
  gemm(stream, t_evec, Wev, t_t4, EK, 64, 64, 64, 64, 64, nullptr, 3, EKC, EKC);
  k_ys<<<gr1(EKS),256,0,stream>>>(t_t1, ns, bsca, knn_cj, EKS);
  k_yv<<<gr1(EKC),256,0,stream>>>(t_t4, t_t2, t_t3, be2n, bn2e, nv, knn_cj, EKC, EKC, (long)NC*64);
  // h = gvl(msg.out, y)
  run_gvl(stream, msg_out, EK, t_t1, t_t4, t_hs, t_hv, t_vh2, t_nrm2, t_g2);

  // q from x (small, lives in freed y_v slot)
  float* t_qs = t_t4; float* t_qv = t_qs + (long)NQ*256;
  float* t_qvh = t_qv + 3L*NQ*64; float* t_qnrm = t_qvh + 3L*NQ*64;
  float* t_qg = t_qnrm + (long)NQ*64;
  run_gvl(stream, pa_q, NQ, xs, xv, t_qs, t_qv, t_qvh, t_qnrm, t_qg);
  // k from h, then logits, then v overwrites k's slots
  float* t_ks = t_vh2; float* t_kv = t_t1;
  run_gvl(stream, pa_k, EK, t_hs, t_hv, t_ks, t_kv, t_evec, t_es, t_g2);
  k_logits1<<<gr1(EK),256,0,stream>>>(t_qs, t_qv, t_ks, t_kv, knn_qi, t_as, t_av,
                                      EK, (long)NQ*64, EKC);
  float* t_vs = t_ks; float* t_vv = t_kv;
  run_gvl(stream, pa_v, EK, t_hs, t_hv, t_vs, t_vv, t_evec, t_es, t_g2);
  // segment softmax (H=1, NQ segments)
  k_fill<<<gr1(NQ),256,0,stream>>>(t_m1, NQ, -INFINITY);
  k_fill<<<gr1(NQ),256,0,stream>>>(t_m2, NQ, -INFINITY);
  k_fill<<<gr1(NQ),256,0,stream>>>(t_sm1, NQ, 0.f);
  k_fill<<<gr1(NQ),256,0,stream>>>(t_sm2, NQ, 0.f);
  k_segmax<<<gr1(EK),256,0,stream>>>(t_as, knn_qi, t_m1, 1, EK);
  k_segmax<<<gr1(EK),256,0,stream>>>(t_av, knn_qi, t_m2, 1, EK);
  k_segexp<<<gr1(EK),256,0,stream>>>(t_as, knn_qi, t_m1, t_sm1, 1, EK);
  k_segexp<<<gr1(EK),256,0,stream>>>(t_av, knn_qi, t_m2, t_sm2, 1, EK);
  k_segnorm<<<gr1(EK),256,0,stream>>>(t_as, knn_qi, t_sm1, 1, EK);
  k_segnorm<<<gr1(EK),256,0,stream>>>(t_av, knn_qi, t_sm2, 1, EK);
  k_fill<<<gr1((long)NQ*256),256,0,stream>>>(t_os, (long)NQ*256, 0.f);
  k_fill<<<gr1(3L*NQ*64),256,0,stream>>>(t_ov, 3L*NQ*64, 0.f);
  k_scat_s<<<gr1(EKS),256,0,stream>>>(t_os, t_vs, t_as, knn_qi, nullptr, 256, 256, 1, EKS);
  k_scat_v<<<gr1(EKC),256,0,stream>>>(t_ov, t_vv, t_av, knn_qi, nullptr, 64, 64, 1, EKC,
                                      (long)NQ*64, EKC);
  run_gvl(stream, pa_out, NQ, t_os, t_ov, ysca, yvec, t_ovh, t_onrm, t_og);

  // ================= PHASE 2: per queried edge features =================
  float* p2 = T;
  float* d_g3  = p2;              float* d_ve3 = d_g3 + EC;
  float* d_s1a = d_ve3 + EV;      float* d_v1a = d_s1a + ES;
  float* d_vh  = d_v1a + EV;      float* d_nr  = d_vh + EV;
  float* d_gg  = d_nr + EC;       float* d_s1  = d_gg + EC;
  float* d_v1  = d_s1 + ES;       float* d_cs  = d_v1 + EV;       // (E,768)
  float* d_cv  = d_cs + (long)E*768;                              // (3,E,192)
  float* d_vhB = d_cv + 3L*(long)E*192;                           // (3,E,192)
  float* d_nrB = d_vhB + 3L*(long)E*192;                          // (E,192)
  float* d_gB  = d_nrB + (long)E*192;
  float* d_s2a = d_gB + EC;       float* d_v2a = d_s2a + ES;
  float* d_vhC = d_v2a + EV;      float* d_nrC = d_vhC + EV;
  float* d_gC  = d_nrC + EC;

  k_edge_geom<<<dim3(E),dim3(64),0,stream>>>(e_iq,e_jc,pos_q,cpx_pos,3.f/63.f,64,64,
      nullptr,0,w_vec3,d_g3,d_ve3,64,E,EC);
  run_gvp(stream, ne0, E, d_g3, d_ve3, d_s1a, d_v1a, d_vh, d_nr, d_gg, d_vh);
  run_gvl(stream, ne1, E, d_s1a, d_v1a, d_s1, d_v1, d_vh, d_nr, d_gg);
  k_cat_s<<<gr1((long)E*768),256,0,stream>>>(d_cs, ysca, node_sca, d_s1, e_iq, e_jc, (long)E*768);
  k_cat_v<<<gr1((long)E*192),256,0,stream>>>(d_cv, yvec, nvT, d_v1, e_iq, e_jc,
      (long)E*192, (long)E*192, (long)NQ*64, (long)NC*64, EC);
  run_gvp(stream, ef0, E, d_cs, d_cv, d_s2a, d_v2a, d_vhB, d_nrB, d_gB, d_vhB);
  run_gvl(stream, ef1, E, d_s2a, d_v2a, s2, v2, d_vhC, d_nrC, d_gC);

  // ================= PHASE 3: attention over edges (H=4) =================
  float* p3 = T;
  float* qs2 = p3;           float* qv2 = qs2 + ES;
  float* ks2 = qv2 + EV;     float* kv2 = ks2 + ES;
  float* vs2 = kv2 + EV;     float* vv2 = vs2 + ES;
  float* vh3 = vv2 + EV;     float* nr3 = vh3 + EV;   float* g3g = nr3 + EC;
  float* bsc = g3g + EC;                          // (PC,256)
  float* bve = bsc + (long)PC*256;                // (3,PC,64)
  float* bvh = bve + 3L*PC*64;                    // (3,PC,64)
  float* bnr = bvh + 3L*PC*64;                    // (PC,64)
  float* bgg = bnr + (long)PC*64;                 // (PC,4)
  float* bv_ = bgg + (long)PC*4;                  // (3,PC,4)
  float* l_s = bv_ + 3L*PC*4;  float* l_v = l_s + Pp*4;
  float* m_s = l_v + Pp*4;     float* m_v = m_s + (long)E*4;
  float* s_s = m_v + (long)E*4; float* s_v = s_s + (long)E*4;
  float* os2 = s_v + (long)E*4; float* ov2 = os2 + ES;

  run_gvl(stream, a_q, E, s2, v2, qs2, qv2, vh3, nr3, g3g);
  run_gvl(stream, a_k, E, s2, v2, ks2, kv2, vh3, nr3, g3g);
  run_gvl(stream, a_v, E, s2, v2, vs2, vv2, vh3, nr3, g3g);
  // triangle-bias, chunked over P
  for (int c = 0; c < 4; ++c) {
    k_edge_geom<<<dim3(PC),dim3(256),0,stream>>>(tri0 + (long)c*PC, tri1 + (long)c*PC,
        cpx_pos, cpx_pos, 10.f/250.f, 251, 256, tri_feat + (long)c*PC*5, 5,
        a_bias_wexp, bsc, bve, 64, PC, (long)PC*64);
    run_gvl(stream, a_bias, PC, bsc, bve, bS + (long)c*PC*4, bv_, bvh, bnr, bgg);
    k_bvsq<<<gr1((long)PC*4),256,0,stream>>>(bv_, bVq + (long)c*PC*4, (long)PC*4, (long)PC*4);
  }
  k_logits4<<<gr1(Pp),256,0,stream>>>(qs2, qv2, ks2, kv2, idx_ei, idx_ej, bS, bVq,
                                      l_s, l_v, Pp, EC);
  k_fill<<<gr1((long)E*4),256,0,stream>>>(m_s, (long)E*4, -INFINITY);
  k_fill<<<gr1((long)E*4),256,0,stream>>>(m_v, (long)E*4, -INFINITY);
  k_fill<<<gr1((long)E*4),256,0,stream>>>(s_s, (long)E*4, 0.f);
  k_fill<<<gr1((long)E*4),256,0,stream>>>(s_v, (long)E*4, 0.f);
  k_segmax<<<gr1(Pp*4),256,0,stream>>>(l_s, idx_ei, m_s, 4, Pp*4);
  k_segmax<<<gr1(Pp*4),256,0,stream>>>(l_v, idx_ei, m_v, 4, Pp*4);
  k_segexp<<<gr1(Pp*4),256,0,stream>>>(l_s, idx_ei, m_s, s_s, 4, Pp*4);
  k_segexp<<<gr1(Pp*4),256,0,stream>>>(l_v, idx_ei, m_v, s_v, 4, Pp*4);
  k_segnorm<<<gr1(Pp*4),256,0,stream>>>(l_s, idx_ei, s_s, 4, Pp*4);
  k_segnorm<<<gr1(Pp*4),256,0,stream>>>(l_v, idx_ei, s_v, 4, Pp*4);
  k_fill<<<gr1(ES),256,0,stream>>>(os2, ES, 0.f);
  k_fill<<<gr1(EV),256,0,stream>>>(ov2, EV, 0.f);
  k_scat_s<<<gr1(Pp*256),256,0,stream>>>(os2, vs2, l_s, idx_ei, idx_ej, 256, 64, 4, Pp*256);
  k_scat_v<<<gr1(Pp*64),256,0,stream>>>(ov2, vv2, l_v, idx_ei, idx_ej, 64, 16, 4, Pp*64, EC, EC);
  k_ln_s<<<dim3(E),dim3(256),0,stream>>>(s2, os2, ln_s_g, ln_s_b);
  k_ln_v<<<dim3(E),dim3(192),0,stream>>>(v2, ov2, ln_v_g, ln_v_b, EC);

  // ================= PHASE 4: affine flow (6 layers) =================
  float* f_hs = T;            float* f_hv = f_hs + ES;
  float* f_vh = f_hv + EV;    float* f_nr = f_vh + EV;
  float* f_g  = f_nr + EC;    float* f_svh = f_g + EC;
  float* f_snr = f_svh + EV;  float* f_so = f_snr + EC;
  float* f_to = f_so + (long)E*4;

  hipMemcpyAsync(zb, z_edge, (size_t)E*4*sizeof(float), hipMemcpyDeviceToDevice, stream);
  k_fill<<<gr1((long)E*4),256,0,stream>>>(ld, (long)E*4, 0.f);
  for (int l = 0; l < 6; ++l) {
    run_gvp(stream, fl_mlp[l], E, s2, v2, f_hs, f_hv, f_vh, f_nr, f_g, f_vh);
    run_gvl(stream, fl_s[l], E, f_hs, f_hv, f_so, nullptr, f_svh, f_snr, nullptr, true);
    run_gvl(stream, fl_t[l], E, f_hs, f_hv, f_to, nullptr, f_svh, f_snr, nullptr, true);
    k_flow<<<gr1((long)E*4),256,0,stream>>>(zb, ld, f_so, f_to, (long)E*4);
  }
}
